// Block_70471823393130
// MI455X (gfx1250) — compile-verified, matrix operand order checked by reading, not run on hardware
//
#include <hip/hip_runtime.h>
#include <math.h>

typedef __bf16 bf16;
typedef __attribute__((ext_vector_type(16))) __bf16 bf16x16;
typedef __attribute__((ext_vector_type(8)))  __bf16 bf16x8;
typedef __attribute__((ext_vector_type(8)))  float  f32x8;
typedef __attribute__((ext_vector_type(4)))  unsigned int u32x4;
typedef __attribute__((ext_vector_type(8)))  int i32x8;
typedef __attribute__((ext_vector_type(4)))  int i32x4;

// Typed address-space pointer casts for CDNA5 async builtins.
#define GP128(p) ((__attribute__((address_space(1))) i32x4*)(p))
#define LP128(p) ((__attribute__((address_space(3))) i32x4*)(p))

__device__ __forceinline__ unsigned lds_offset_of(const void* p) {
  return (unsigned)(unsigned long long)(const __attribute__((address_space(3))) void*)p;
}

__device__ __forceinline__ void wait_async0() {
#if __has_builtin(__builtin_amdgcn_s_wait_asynccnt)
  __builtin_amdgcn_s_wait_asynccnt(0);
#else
  asm volatile("s_wait_asynccnt 0" ::: "memory");
#endif
}

#if __has_builtin(__builtin_amdgcn_tensor_load_to_lds)
#define HAS_TDM 1
// 2D TDM load: tile (32 x rows) of bf16, global row stride = strideElems,
// LDS rows padded 16 data DWORDs + 4 pad DWORDs -> ld 40 elements.
__device__ __forceinline__ void tdm_load_tile(unsigned lds_addr, const void* gptr,
                                              unsigned rows, unsigned strideElems) {
  unsigned long long ga = (unsigned long long)gptr;
  u32x4 g0;
  g0[0] = 1u;                                   // count=1, user mode
  g0[1] = lds_addr;                             // lds_addr
  g0[2] = (unsigned)ga;                         // global_addr[31:0]
  g0[3] = (unsigned)((ga >> 32) & 0x01FFFFFFu)  // global_addr[56:32]
          | (2u << 30);                         // type=2 ("image")
  i32x8 g1;
  g1[0] = (int)((1u << 16)      // data_size = 2 bytes
                | (1u << 20)    // pad_enable
                | (3u << 22)    // pad_interval: 16 DWORDs (one 32-elem row)
                | (3u << 25));  // pad_amount: 4 DWORDs (8 bf16) -> ld 40
  g1[1] = (int)((strideElems & 0xFFFFu) << 16);          // tensor_dim0 lo
  g1[2] = (int)(((strideElems >> 16) & 0xFFFFu)          // tensor_dim0 hi
                | ((rows & 0xFFFFu) << 16));             // tensor_dim1 lo
  g1[3] = (int)(32u << 16);       // tensor_dim1 hi = 0, tile_dim0 = 32
  g1[4] = (int)rows;              // tile_dim1 = rows, tile_dim2 = 0
  g1[5] = (int)strideElems;       // tensor_dim0_stride lo
  g1[6] = 0;                      // stride hi / dim1_stride lo
  g1[7] = 0;
  i32x4 gz4 = {0, 0, 0, 0};
  i32x8 gz8 = {0, 0, 0, 0, 0, 0, 0, 0};
  __builtin_amdgcn_tensor_load_to_lds(g0, g1, gz4, gz4, gz8, 0);
}
#else
#define HAS_TDM 0
#endif

// ---------------------------------------------------------------------------
// 16-bit A-fragment loader for V_WMMA_*_16X16X32_BF16 (ISA 7.12.2).
// Serves the B side too when the tile is stored transposed (Bt[n][k]).
// ---------------------------------------------------------------------------
__device__ __forceinline__ bf16x16 ld_frag(const bf16* base, int ld, int row0,
                                           int k0, int lane) {
  int r  = row0 + (lane & 15);
  int kh = (lane >> 4) << 3;
  const bf16* p = base + r * ld + k0 + kh;
  bf16x8 lo = *(const bf16x8*)(p);
  bf16x8 hi = *(const bf16x8*)(p + 16);
  bf16x16 out;
#pragma unroll
  for (int i = 0; i < 8; ++i) { out[i] = lo[i]; out[i + 8] = hi[i]; }
  return out;
}

// ---------------------------------------------------------------------------
// f32 [K][N] -> bf16 [N][K] transposed cast (weights), 32x32 LDS tiles.
// ---------------------------------------------------------------------------
__global__ __launch_bounds__(256) void k_cast_transpose_bf16(
    const float* __restrict__ src, bf16* __restrict__ dst, int K, int N) {
  __shared__ float t[32][33];
  int k0 = blockIdx.x * 32, n0 = blockIdx.y * 32;
  int tx = threadIdx.x & 31, ty = threadIdx.x >> 5;  // 32 x 8
#pragma unroll
  for (int i = 0; i < 4; ++i)
    t[ty + i * 8][tx] = src[(long)(k0 + ty + i * 8) * N + n0 + tx];
  __syncthreads();
#pragma unroll
  for (int i = 0; i < 4; ++i)
    dst[(long)(n0 + ty + i * 8) * K + k0 + tx] = (bf16)t[tx][ty + i * 8];
}

// ---------------------------------------------------------------------------
__global__ void k_rmsnorm_bf16(const float* __restrict__ x, const float* __restrict__ w,
                               bf16* __restrict__ out, int C) {
  int row = blockIdx.x;
  const float* xr = x + (long)row * C;
  bf16* orow = out + (long)row * C;
  float ss = 0.f;
  for (int c = threadIdx.x; c < C; c += blockDim.x) { float v = xr[c]; ss += v * v; }
  __shared__ float red[256];
  red[threadIdx.x] = ss;
  __syncthreads();
  for (int s = 128; s > 0; s >>= 1) {
    if ((int)threadIdx.x < s) red[threadIdx.x] += red[threadIdx.x + s];
    __syncthreads();
  }
  float rms = rsqrtf(red[0] / (float)C + 1e-6f);
  for (int c = threadIdx.x; c < C; c += blockDim.x)
    orow[c] = (bf16)(w[c] * xr[c] * rms);
}

// ---------------------------------------------------------------------------
// bf16 WMMA GEMM: Out[M,N] = A[M,K] @ Bt[N,K]^T  (A row-major, B transposed)
// Block tile 128(M) x 256(N), BK=32, 8 waves (2x4), wave tile 64x64 (4x4).
// Both tiles staged by the Tensor Data Mover into double-buffered LDS,
// pipelined with s_wait_tensorcnt<=2 (2 descriptors in flight per buffer).
// EPI: 0 = f32, 1 = bf16, 2 = f32 (acc + Res)
// ---------------------------------------------------------------------------
template <int EPI>
__global__ __launch_bounds__(256) void k_gemm_bf16(
    const bf16* __restrict__ A, const bf16* __restrict__ Bt,
    void* __restrict__ Out, const float* __restrict__ Res,
    int M, int N, int K) {
  __shared__ __attribute__((aligned(16))) bf16 sA[2 * 128 * 40];   // A[m][k]
  __shared__ __attribute__((aligned(16))) bf16 sB[2 * 256 * 40];   // Bt[n][k]
  const int tid = threadIdx.x;
  const int lane = tid & 31, wid = tid >> 5;
  const int wm = wid & 1, wn = wid >> 1;
  const int m0 = blockIdx.y * 128, n0 = blockIdx.x * 256;

  f32x8 acc[4][4];
#pragma unroll
  for (int a = 0; a < 4; ++a)
#pragma unroll
    for (int b = 0; b < 4; ++b)
#pragma unroll
      for (int i = 0; i < 8; ++i) acc[a][b][i] = 0.f;

#if HAS_TDM
  const unsigned ldsA_off = lds_offset_of(sA);
  const unsigned ldsB_off = lds_offset_of(sB);
#endif

  // stage tile `ks` into buffer `buf`
  auto stage = [&](int ks, int buf) {
    int k0 = ks * 32;
#if HAS_TDM
    if (wid == 0) {
      tdm_load_tile(ldsA_off + (unsigned)buf * 128 * 40 * 2,
                    &A[(long)m0 * K + k0], 128, (unsigned)K);
      tdm_load_tile(ldsB_off + (unsigned)buf * 256 * 40 * 2,
                    &Bt[(long)n0 * K + k0], 256, (unsigned)K);
    }
#else
    bf16* sAb = sA + buf * 128 * 40;
    bf16* sBb = sB + buf * 256 * 40;
#pragma unroll
    for (int it = 0; it < 2; ++it) {
      int chunk = it * 256 + tid;
      int r = chunk >> 2, c8 = (chunk & 3) << 3;
      *(bf16x8*)&sAb[r * 40 + c8] =
          *(const bf16x8*)&A[(long)(m0 + r) * K + k0 + c8];
    }
#pragma unroll
    for (int it = 0; it < 4; ++it) {
      int chunk = it * 256 + tid;
      int r = chunk >> 2, c8 = (chunk & 3) << 3;
      *(bf16x8*)&sBb[r * 40 + c8] =
          *(const bf16x8*)&Bt[(long)(n0 + r) * K + k0 + c8];
    }
#endif
  };

  const int nk = K / 32;
  stage(0, 0);
  for (int ks = 0; ks < nk; ++ks) {
    const int cur = ks & 1;
    const bool more = (ks + 1) < nk;
    __syncthreads();  // close out compute on the buffer we are about to refill
    if (more) stage(ks + 1, cur ^ 1);
    if (more && ks + 2 < nk) {  // warm L2 two tiles ahead (global_prefetch_b8)
      __builtin_prefetch(&A[(long)(m0 + (tid >> 1)) * K + (ks + 2) * 32], 0, 0);
      __builtin_prefetch(&Bt[(long)(n0 + tid) * K + (ks + 2) * 32], 0, 0);
    }
#if HAS_TDM
    if (wid == 0) {
      if (more) __builtin_amdgcn_s_wait_tensorcnt((unsigned short)2);
      else      __builtin_amdgcn_s_wait_tensorcnt((unsigned short)0);
    }
#endif
    __syncthreads();  // publish buffer `cur`

    const bf16* sAb = sA + cur * 128 * 40;
    const bf16* sBb = sB + cur * 256 * 40;
    bf16x16 af[4], bfr[4];
#pragma unroll
    for (int i = 0; i < 4; ++i)
      af[i] = ld_frag(sAb, 40, wm * 64 + i * 16, 0, lane);
#pragma unroll
    for (int j = 0; j < 4; ++j)
      bfr[j] = ld_frag(sBb, 40, wn * 64 + j * 16, 0, lane);
#pragma unroll
    for (int i = 0; i < 4; ++i)
#pragma unroll
      for (int j = 0; j < 4; ++j)
        acc[i][j] = __builtin_amdgcn_wmma_f32_16x16x32_bf16(
            false, af[i], false, bfr[j], (short)0, acc[i][j], false, false);
  }

  const int rh = (lane >> 4) << 3, cl = lane & 15;
#pragma unroll
  for (int mi = 0; mi < 4; ++mi)
#pragma unroll
    for (int j = 0; j < 4; ++j) {
      int col = n0 + wn * 64 + j * 16 + cl;
#pragma unroll
      for (int i = 0; i < 8; ++i) {
        int row = m0 + wm * 64 + mi * 16 + rh + i;
        long off = (long)row * N + col;
        float v = acc[mi][j][i];
        if (EPI == 0)      ((float*)Out)[off] = v;
        else if (EPI == 1) ((bf16*)Out)[off]  = (bf16)v;
        else               ((float*)Out)[off] = v + Res[off];
      }
    }
}

// ---------------------------------------------------------------------------
// RoPE + split qkv[b,t,3C] -> q,k,v in [b,h,t,d] (bf16), rope on q,k
// ---------------------------------------------------------------------------
__global__ void k_rope_split(const bf16* __restrict__ qkv, bf16* __restrict__ q,
                             bf16* __restrict__ k, bf16* __restrict__ v,
                             int B, int T, int H, int Dh) {
  long idx = (long)blockIdx.x * blockDim.x + threadIdx.x;
  long total = (long)B * T * H * 64;
  if (idx >= total) return;
  int j = (int)(idx & 63);
  long r = idx >> 6;
  int h = (int)(r % H); r /= H;
  int t = (int)(r % T); int b = (int)(r / T);
  long mrow = ((long)b * T + t) * (3L * H * Dh);
  long qo = mrow + (long)h * Dh;
  long ko = mrow + (long)H * Dh + (long)h * Dh;
  long vo = mrow + 2L * H * Dh + (long)h * Dh;
  float q1 = (float)qkv[qo + j], q2 = (float)qkv[qo + 64 + j];
  float k1 = (float)qkv[ko + j], k2 = (float)qkv[ko + 64 + j];
  float v1 = (float)qkv[vo + j], v2 = (float)qkv[vo + 64 + j];
  float freq = __powf(10000.f, -(float)j * (1.f / 64.f));
  float ang = (float)t * freq, sn, cs;
  __sincosf(ang, &sn, &cs);
  long ob = (((long)b * H + h) * T + t) * Dh;
  q[ob + j]      = (bf16)(q1 * cs - q2 * sn);
  q[ob + 64 + j] = (bf16)(q1 * sn + q2 * cs);
  k[ob + j]      = (bf16)(k1 * cs - k2 * sn);
  k[ob + 64 + j] = (bf16)(k1 * sn + k2 * cs);
  v[ob + j]      = (bf16)v1;
  v[ob + 64 + j] = (bf16)v2;
}

// ---------------------------------------------------------------------------
// Flash attention (causal), bf16 WMMA, D=128.
// grid = (T/128, B*H); 256 threads (8 waves), wave owns 16 q-rows.
// K-tile (64 keys) staged via async global->LDS (ASYNCcnt path).
// ---------------------------------------------------------------------------
__global__ __launch_bounds__(256) void k_flash_attn(
    const bf16* __restrict__ Q, const bf16* __restrict__ Km,
    const bf16* __restrict__ Vm, bf16* __restrict__ Out,
    int B, int H, int T) {
  __shared__ __attribute__((aligned(16))) bf16 sK[64 * 136];    // K[key][d]
  __shared__ __attribute__((aligned(16))) bf16 sVt[128 * 72];   // Vt[d][key]
  __shared__ __attribute__((aligned(16))) bf16 sP[8 * 16 * 72]; // per-wave P
  const int tid = threadIdx.x, lane = tid & 31, wid = tid >> 5;
  const int qb = blockIdx.x, bh = blockIdx.y;
  const int b = bh / H, h = bh % H;
  const int q0 = qb * 128;
  const long base = (long)bh * T * 128;
  const float scale = 0.08838834764831845f;  // 1/sqrt(128)
  const int cl = lane & 15, rh = (lane >> 4) << 3;

  bf16x16 aq[4];
  {
    const bf16* qp = Q + base + (long)(q0 + wid * 16 + cl) * 128 + rh;
#pragma unroll
    for (int kk = 0; kk < 4; ++kk) {
      bf16x8 lo = *(const bf16x8*)(qp + kk * 32);
      bf16x8 hi = *(const bf16x8*)(qp + kk * 32 + 16);
#pragma unroll
      for (int i = 0; i < 8; ++i) { aq[kk][i] = lo[i]; aq[kk][i + 8] = hi[i]; }
    }
  }

  f32x8 o[8];
#pragma unroll
  for (int d = 0; d < 8; ++d)
#pragma unroll
    for (int i = 0; i < 8; ++i) o[d][i] = 0.f;
  float m_i[8], l_i[8];
#pragma unroll
  for (int i = 0; i < 8; ++i) { m_i[i] = -INFINITY; l_i[i] = 0.f; }

  const int nkt = qb * 2 + 2;  // causal: keys < (qb+1)*128
  for (int kb = 0; kb < nkt; ++kb) {
    __syncthreads();
#pragma unroll
    for (int it = 0; it < 4; ++it) {
      int chunk = it * 256 + tid;
      int r = chunk >> 4, c8 = (chunk & 15) << 3;
#if __has_builtin(__builtin_amdgcn_global_load_async_to_lds_b128)
      __builtin_amdgcn_global_load_async_to_lds_b128(
          GP128(&Km[base + (long)(kb * 64 + r) * 128 + c8]),
          LP128(&sK[r * 136 + c8]), 0, 0);
#else
      *(bf16x8*)&sK[r * 136 + c8] =
          *(const bf16x8*)&Km[base + (long)(kb * 64 + r) * 128 + c8];
#endif
      bf16x8 vv = *(const bf16x8*)&Vm[base + (long)(kb * 64 + r) * 128 + c8];
#pragma unroll
      for (int j = 0; j < 8; ++j) sVt[(c8 + j) * 72 + r] = vv[j];
    }
#if __has_builtin(__builtin_amdgcn_global_load_async_to_lds_b128)
    wait_async0();
#endif
    __syncthreads();

    f32x8 s[4];
#pragma unroll
    for (int nf = 0; nf < 4; ++nf) {
#pragma unroll
      for (int i = 0; i < 8; ++i) s[nf][i] = 0.f;
#pragma unroll
      for (int kk = 0; kk < 4; ++kk) {
        bf16x16 bk = ld_frag(sK, 136, nf * 16, kk * 32, lane);
        s[nf] = __builtin_amdgcn_wmma_f32_16x16x32_bf16(
            false, aq[kk], false, bk, (short)0, s[nf], false, false);
      }
    }

    float tm[8];
#pragma unroll
    for (int i = 0; i < 8; ++i) tm[i] = -INFINITY;
#pragma unroll
    for (int nf = 0; nf < 4; ++nf) {
      int col = kb * 64 + nf * 16 + cl;
#pragma unroll
      for (int i = 0; i < 8; ++i) {
        int row = q0 + wid * 16 + rh + i;
        float val = s[nf][i] * scale;
        if (col > row) val = -INFINITY;
        s[nf][i] = val;
        tm[i] = fmaxf(tm[i], val);
      }
    }
#pragma unroll
    for (int off = 1; off < 16; off <<= 1)
#pragma unroll
      for (int i = 0; i < 8; ++i) tm[i] = fmaxf(tm[i], __shfl_xor(tm[i], off, 32));

    float alpha[8], rs[8];
#pragma unroll
    for (int i = 0; i < 8; ++i) {
      float mn = fmaxf(m_i[i], tm[i]);
      alpha[i] = __expf(m_i[i] - mn);
      m_i[i] = mn;
      rs[i] = 0.f;
    }
    bf16* pw = sP + wid * 16 * 72;
#pragma unroll
    for (int nf = 0; nf < 4; ++nf)
#pragma unroll
      for (int i = 0; i < 8; ++i) {
        float p = __expf(s[nf][i] - m_i[i]);
        rs[i] += p;
        pw[(rh + i) * 72 + nf * 16 + cl] = (bf16)p;
      }
#pragma unroll
    for (int off = 1; off < 16; off <<= 1)
#pragma unroll
      for (int i = 0; i < 8; ++i) rs[i] += __shfl_xor(rs[i], off, 32);
#pragma unroll
    for (int i = 0; i < 8; ++i) l_i[i] = l_i[i] * alpha[i] + rs[i];
#pragma unroll
    for (int d = 0; d < 8; ++d)
#pragma unroll
      for (int i = 0; i < 8; ++i) o[d][i] *= alpha[i];

    bf16x16 ap[2];
#pragma unroll
    for (int kk = 0; kk < 2; ++kk) ap[kk] = ld_frag(pw, 72, 0, kk * 32, lane);
#pragma unroll
    for (int df = 0; df < 8; ++df)
#pragma unroll
      for (int kk = 0; kk < 2; ++kk) {
        bf16x16 bv = ld_frag(sVt, 72, df * 16, kk * 32, lane);
        o[df] = __builtin_amdgcn_wmma_f32_16x16x32_bf16(
            false, ap[kk], false, bv, (short)0, o[df], false, false);
      }
  }

#pragma unroll
  for (int i = 0; i < 8; ++i) {
    int row = q0 + wid * 16 + rh + i;
    float inv = 1.f / l_i[i];
#pragma unroll
    for (int df = 0; df < 8; ++df)
      Out[((long)b * T + row) * (long)(H * 128) + h * 128 + df * 16 + cl] =
          (bf16)(o[df][i] * inv);
  }
}

// ---------------------------------------------------------------------------
__global__ void k_silu_mul(bf16* __restrict__ g, const bf16* __restrict__ u, long n) {
  long i = (long)blockIdx.x * blockDim.x + threadIdx.x;
  long stride = (long)gridDim.x * blockDim.x;
  for (; i < n; i += stride) {
    float a = (float)g[i];
    float bb = (float)u[i];
    float s = a / (1.f + __expf(-a));
    g[i] = (bf16)(s * bb);
  }
}

// ---------------------------------------------------------------------------
extern "C" void kernel_launch(void* const* d_in, const int* in_sizes, int n_in,
                              void* d_out, int out_size, void* d_ws, size_t ws_size,
                              hipStream_t stream) {
  (void)in_sizes; (void)n_in; (void)out_size; (void)ws_size;
  const long Bb = 2, T = 2048, C = 2048, C3 = 6144, Hid = 5632;
  const int H = 16;
  const long M = Bb * T;

  const float* x       = (const float*)d_in[0];
  const float* w_norm1 = (const float*)d_in[1];
  const float* w_qkv   = (const float*)d_in[2];
  const float* w_proj  = (const float*)d_in[3];
  const float* w_norm2 = (const float*)d_in[4];
  const float* w1      = (const float*)d_in[5];
  const float* w2      = (const float*)d_in[6];
  const float* w3      = (const float*)d_in[7];

  char* ws = (char*)d_ws;
  auto alloc = [&](long bytes) { char* p = ws; ws += (bytes + 255) & ~255L; return p; };
  // all weights stored TRANSPOSED: [N][K] bf16
  bf16* wqkv_t  = (bf16*)alloc(C * C3 * 2);
  bf16* wproj_t = (bf16*)alloc(C * C * 2);
  bf16* w1_t    = (bf16*)alloc(C * Hid * 2);
  bf16* w2_t    = (bf16*)alloc(C * Hid * 2);
  bf16* w3_t    = (bf16*)alloc(Hid * C * 2);
  bf16* hbf     = (bf16*)alloc(M * C * 2);             // h, later h2
  char* big     = alloc(M * C3 * 2 + 3 * M * C * 2);   // qkv+q/k/v, later g/u
  bf16* qkv_bf = (bf16*)big;
  bf16* q_bf   = (bf16*)(big + M * C3 * 2);
  bf16* k_bf   = q_bf + M * C;
  bf16* v_bf   = k_bf + M * C;
  bf16* attn_bf = (bf16*)alloc(M * C * 2);
  float* x2     = (float*)alloc(M * C * 4);
  bf16* g_bf = (bf16*)big;  // aliases qkv region (free after attention)
  bf16* u_bf = g_bf + M * Hid;

  // 1. transpose-cast weights to bf16 [N][K]
  {
    dim3 b(256);
    k_cast_transpose_bf16<<<dim3(C / 32, C3 / 32), b, 0, stream>>>(w_qkv, wqkv_t, (int)C, (int)C3);
    k_cast_transpose_bf16<<<dim3(C / 32, C / 32), b, 0, stream>>>(w_proj, wproj_t, (int)C, (int)C);
    k_cast_transpose_bf16<<<dim3(C / 32, Hid / 32), b, 0, stream>>>(w1, w1_t, (int)C, (int)Hid);
    k_cast_transpose_bf16<<<dim3(C / 32, Hid / 32), b, 0, stream>>>(w2, w2_t, (int)C, (int)Hid);
    k_cast_transpose_bf16<<<dim3(Hid / 32, C / 32), b, 0, stream>>>(w3, w3_t, (int)Hid, (int)C);
  }

  k_rmsnorm_bf16<<<(int)M, 256, 0, stream>>>(x, w_norm1, hbf, (int)C);

  // qkv = h @ w_qkv
  dim3 g1((unsigned)(C3 / 256), (unsigned)(M / 128));
  k_gemm_bf16<1><<<g1, 256, 0, stream>>>(hbf, wqkv_t, qkv_bf, nullptr,
                                         (int)M, (int)C3, (int)C);
  long nrope = Bb * T * H * 64;
  k_rope_split<<<(unsigned)((nrope + 255) / 256), 256, 0, stream>>>(
      qkv_bf, q_bf, k_bf, v_bf, (int)Bb, (int)T, H, 128);

  dim3 ga((unsigned)(T / 128), (unsigned)(Bb * H));
  k_flash_attn<<<ga, 256, 0, stream>>>(q_bf, k_bf, v_bf, attn_bf,
                                       (int)Bb, H, (int)T);

  // x2 = x + attn @ w_proj
  dim3 g2((unsigned)(C / 256), (unsigned)(M / 128));
  k_gemm_bf16<2><<<g2, 256, 0, stream>>>(attn_bf, wproj_t, x2, x,
                                         (int)M, (int)C, (int)C);
  k_rmsnorm_bf16<<<(int)M, 256, 0, stream>>>(x2, w_norm2, hbf, (int)C);

  // g = h2@w1, u = h2@w2
  dim3 g3((unsigned)(Hid / 256), (unsigned)(M / 128));
  k_gemm_bf16<1><<<g3, 256, 0, stream>>>(hbf, w1_t, g_bf, nullptr,
                                         (int)M, (int)Hid, (int)C);
  k_gemm_bf16<1><<<g3, 256, 0, stream>>>(hbf, w2_t, u_bf, nullptr,
                                         (int)M, (int)Hid, (int)C);
  k_silu_mul<<<4096, 256, 0, stream>>>(g_bf, u_bf, M * Hid);

  // out = x2 + g @ w3
  dim3 g4((unsigned)(C / 256), (unsigned)(M / 128));
  k_gemm_bf16<2><<<g4, 256, 0, stream>>>(g_bf, w3_t, (float*)d_out, x2,
                                         (int)M, (int)C, (int)Hid);
}
